// SelfAttentionLayer_14757507629104
// MI455X (gfx1250) — compile-verified, matrix operand order checked by reading, not run on hardware
//
#include <hip/hip_runtime.h>
#include <hip/hip_bf16.h>

typedef __attribute__((ext_vector_type(16))) _Float16 v16h;
typedef __attribute__((ext_vector_type(8)))  float    v8f;

#define Bdim 16
#define Ldim 1024
#define Ddim 256
#define Hdim 8
#define DHdim 32
#define FFdim 2048
#define BLdim (Bdim * Ldim)

// ---------------------------------------------------------------------------
// helpers
// ---------------------------------------------------------------------------
__device__ __forceinline__ v8f zero8() {
  v8f z;
#pragma unroll
  for (int i = 0; i < 8; ++i) z[i] = 0.0f;
  return z;
}

// Load a 16x32 (f16) WMMA A/B fragment for this lane from a row pointer.
// Per CDNA5 ISA 7.12.2: lane r=lane%16 owns row r; half=lane/16 selects
// K chunks [8h..8h+7] and [16+8h..23+8h] -> two 16B loads.
__device__ __forceinline__ v16h load_frag(const _Float16* __restrict__ rowp, int half) {
  v16h f;
  reinterpret_cast<float4*>(&f)[0] = *reinterpret_cast<const float4*>(rowp + 8 * half);
  reinterpret_cast<float4*>(&f)[1] = *reinterpret_cast<const float4*>(rowp + 16 + 8 * half);
  return f;
}

__device__ __forceinline__ v8f wmma_f16(v16h a, v16h b, v8f c) {
  return __builtin_amdgcn_wmma_f32_16x16x32_f16(false, a, false, b, (short)0, c, false, false);
}

// CDNA5 async memory->LDS copy (16B per lane), tracked by ASYNCcnt.
// vdst = VGPR with LDS byte address (low 32 bits of the generic pointer),
// vaddr = 64-bit global address.
__device__ __forceinline__ void async_copy_b128(const _Float16* g, _Float16* lds) {
  unsigned lo = (unsigned)(unsigned long long)(uintptr_t)lds;
  asm volatile("global_load_async_to_lds_b128 %0, %1, off"
               :: "v"(lo), "v"(g)
               : "memory");
}
#define WAIT_ASYNCCNT(n) asm volatile("s_wait_asynccnt " #n ::: "memory")

// ---------------------------------------------------------------------------
// conversion kernels
// ---------------------------------------------------------------------------
__global__ __launch_bounds__(256) void cvt_f32_to_f16(const float* __restrict__ in,
                                                      _Float16* __restrict__ out, int n) {
  int i = blockIdx.x * 256 + threadIdx.x;
  if (i < n) out[i] = (_Float16)in[i];
}

// Wt[n*K + k] = (f16) W[k*N + n]   (store weights as [N,K] so B loads like A)
__global__ __launch_bounds__(256) void transpose_cvt(const float* __restrict__ W,
                                                     _Float16* __restrict__ Wt,
                                                     int K, int N) {
  int i = blockIdx.x * 256 + threadIdx.x;
  if (i >= N * K) return;
  int n = i / K, k = i - n * K;
  Wt[i] = (_Float16)W[(size_t)k * N + n];
}

// ---------------------------------------------------------------------------
// generic WMMA GEMM: C[M,N] = A[M,K] @ Bt[N,K]^T (+bias) with fused epilogues
// block: 256 threads = 8 waves (4 M x 2 N), tile 128x128, K-step 32.
// Staging: double-buffered GLOBAL_LOAD_ASYNC_TO_LDS_B128 (ASYNCcnt-tracked),
// next slab's loads in flight behind the current slab's 8 WMMAs.
// ---------------------------------------------------------------------------
#define GEMM_BM 128
#define GEMM_BN 128
#define GEMM_BK 32
#define LDSS 40  // padded row stride in halves (80B) to avoid bank conflicts
#define BUFH (GEMM_BM * LDSS)

enum { MODE_F16_PLAIN = 0, MODE_F16_HEADS = 1, MODE_F16_HEADST = 2,
       MODE_F16_RELU = 3, MODE_F32_PLAIN = 4 };

__global__ __launch_bounds__(256) void gemm_wmma(const _Float16* __restrict__ A,
                                                 const _Float16* __restrict__ Bt,
                                                 const float* __restrict__ bias,
                                                 void* __restrict__ Cout,
                                                 int M, int N, int K,
                                                 int mode, float scale) {
  __shared__ __align__(16) _Float16 As[2 * BUFH];
  __shared__ __align__(16) _Float16 Bs[2 * BUFH];

  const int tid = threadIdx.x;
  const int wave = tid >> 5, lane = tid & 31;
  const int wm = wave & 3, wn = wave >> 2;   // 4 x 2 wave grid
  const int r = lane & 15, hl = lane >> 4;
  const int m0 = blockIdx.x * GEMM_BM;
  const int n0 = blockIdx.y * GEMM_BN;

  // per-thread staging coordinates (4 async b128 per wave per slab)
  const int row0 = tid >> 2, col8a = (tid & 3) * 8;
  const int row1 = row0 + 64;

  auto stage = [&](int kbase, int buf) {
    _Float16* Ab = As + buf * BUFH;
    _Float16* Bb = Bs + buf * BUFH;
    async_copy_b128(A  + (size_t)(m0 + row0) * K + kbase + col8a, Ab + row0 * LDSS + col8a);
    async_copy_b128(Bt + (size_t)(n0 + row0) * K + kbase + col8a, Bb + row0 * LDSS + col8a);
    async_copy_b128(A  + (size_t)(m0 + row1) * K + kbase + col8a, Ab + row1 * LDSS + col8a);
    async_copy_b128(Bt + (size_t)(n0 + row1) * K + kbase + col8a, Bb + row1 * LDSS + col8a);
  };

  v8f c[2][4];
#pragma unroll
  for (int mi = 0; mi < 2; ++mi)
#pragma unroll
    for (int ni = 0; ni < 4; ++ni) c[mi][ni] = zero8();

  const int ksteps = K / GEMM_BK;
  stage(0, 0);

  for (int kt = 0; kt < ksteps; ++kt) {
    const int buf = kt & 1;
    if (kt + 1 < ksteps) {
      stage((kt + 1) * GEMM_BK, buf ^ 1);
      WAIT_ASYNCCNT(4);   // in-order: current slab's 4 loads have landed
    } else {
      WAIT_ASYNCCNT(0);
    }
    __syncthreads();      // all waves' current-slab data visible

    const _Float16* Ab = As + buf * BUFH;
    const _Float16* Bb = Bs + buf * BUFH;
    v16h af[2], bf[4];
#pragma unroll
    for (int mi = 0; mi < 2; ++mi)
      af[mi] = load_frag(Ab + (wm * 32 + mi * 16 + r) * LDSS, hl);
#pragma unroll
    for (int ni = 0; ni < 4; ++ni)
      bf[ni] = load_frag(Bb + (wn * 64 + ni * 16 + r) * LDSS, hl);

#pragma unroll
    for (int mi = 0; mi < 2; ++mi)
#pragma unroll
      for (int ni = 0; ni < 4; ++ni)
        c[mi][ni] = wmma_f16(af[mi], bf[ni], c[mi][ni]);
    __syncthreads();      // buffer free before it is overwritten next round
  }

  // epilogue: C layout -> lane n = lane%16, element j has row j + 8*(lane/16)
#pragma unroll
  for (int mi = 0; mi < 2; ++mi) {
#pragma unroll
    for (int ni = 0; ni < 4; ++ni) {
      const int col = n0 + wn * 64 + ni * 16 + r;
      const float bv = bias[col];
#pragma unroll
      for (int j = 0; j < 8; ++j) {
        const int row = m0 + wm * 32 + mi * 16 + j + 8 * hl;
        float v = (c[mi][ni][j] + bv) * scale;
        if (mode == MODE_F16_PLAIN) {
          ((_Float16*)Cout)[(size_t)row * N + col] = (_Float16)v;
        } else if (mode == MODE_F16_RELU) {
          v = fmaxf(v, 0.0f);
          ((_Float16*)Cout)[(size_t)row * N + col] = (_Float16)v;
        } else if (mode == MODE_F32_PLAIN) {
          ((float*)Cout)[(size_t)row * N + col] = v;
        } else {
          // head scatter: row = b*L + l, col = h*DH + dh
          int bb = row >> 10, ll = row & (Ldim - 1);
          int hh = col >> 5, dd = col & (DHdim - 1);
          if (mode == MODE_F16_HEADS) {
            ((_Float16*)Cout)[(((size_t)(bb * Hdim + hh)) * Ldim + ll) * DHdim + dd] = (_Float16)v;
          } else { // MODE_F16_HEADST : [B,H,DH,L]
            ((_Float16*)Cout)[(((size_t)(bb * Hdim + hh)) * DHdim + dd) * Ldim + ll] = (_Float16)v;
          }
        }
      }
    }
  }
}

// ---------------------------------------------------------------------------
// flash attention: one block = 4 waves, each wave owns 16 query rows of one
// (b,h). DH=32 == WMMA K, so Q is a single A fragment held in registers.
// grid: B*H*(L/64)
// ---------------------------------------------------------------------------
__global__ __launch_bounds__(128) void flash_attn(const _Float16* __restrict__ qh,
                                                  const _Float16* __restrict__ kh,
                                                  const _Float16* __restrict__ vt,
                                                  _Float16* __restrict__ o16) {
  __shared__ float s_tile[4][16][33];
  __shared__ __align__(16) _Float16 p_tile[4][16][LDSS];
  __shared__ float mrow[4][16], lrow[4][16], arow[4][16];

  const int wave = threadIdx.x >> 5, lane = threadIdx.x & 31;
  const int r = lane & 15, hl = lane >> 4;
  const int qblk = blockIdx.x & (Ldim / 64 - 1);
  const int bh = blockIdx.x >> 4;          // L/64 = 16 blocks per (b,h)
  const int q0 = qblk * 64 + wave * 16;

  const _Float16* qbase = qh + (size_t)bh * Ldim * DHdim;
  const _Float16* kbase = kh + (size_t)bh * Ldim * DHdim;
  const _Float16* vbase = vt + (size_t)bh * DHdim * Ldim;

  const v16h aq = load_frag(qbase + (size_t)(q0 + r) * DHdim, hl);
  v8f o0 = zero8(), o1 = zero8();

  if (lane < 16) { mrow[wave][lane] = -1e30f; lrow[wave][lane] = 0.0f; }

  for (int kc = 0; kc < Ldim / 32; ++kc) {
    // S = (Q*scale) @ K^T   (scale folded into Q at projection time)
    v16h bk0 = load_frag(kbase + (size_t)(kc * 32 + r) * DHdim, hl);
    v16h bk1 = load_frag(kbase + (size_t)(kc * 32 + 16 + r) * DHdim, hl);
    v8f s0 = wmma_f16(aq, bk0, zero8());
    v8f s1 = wmma_f16(aq, bk1, zero8());
#pragma unroll
    for (int j = 0; j < 8; ++j) {
      s_tile[wave][j + 8 * hl][r]      = s0[j];
      s_tile[wave][j + 8 * hl][16 + r] = s1[j];
    }

    // online softmax (lanes 0-15, one row each) -- same-wave LDS is in order
    if (lane < 16) {
      const int row = lane;
      float mo = mrow[wave][row];
      float mx = mo;
      float sv[32];
#pragma unroll
      for (int cidx = 0; cidx < 32; ++cidx) {
        sv[cidx] = s_tile[wave][row][cidx];
        mx = fmaxf(mx, sv[cidx]);
      }
      float sum = 0.0f;
#pragma unroll
      for (int cidx = 0; cidx < 32; ++cidx) {
        float p = __expf(sv[cidx] - mx);
        sum += p;
        p_tile[wave][row][cidx] = (_Float16)p;
      }
      float al = __expf(mo - mx);
      arow[wave][row] = al;
      mrow[wave][row] = mx;
      lrow[wave][row] = al * lrow[wave][row] + sum;
    }

    // rescale O and accumulate P @ V
    v16h ap  = load_frag(&p_tile[wave][r][0], hl);
    v16h bv0 = load_frag(vbase + (size_t)r * Ldim + kc * 32, hl);
    v16h bv1 = load_frag(vbase + (size_t)(16 + r) * Ldim + kc * 32, hl);
#pragma unroll
    for (int j = 0; j < 8; ++j) {
      float al = arow[wave][j + 8 * hl];
      o0[j] *= al;
      o1[j] *= al;
    }
    o0 = wmma_f16(ap, bv0, o0);
    o1 = wmma_f16(ap, bv1, o1);
  }

  // write O back to [B*L, D] f16, normalized by row sums
  const int b = bh >> 3, h = bh & (Hdim - 1);
#pragma unroll
  for (int j = 0; j < 8; ++j) {
    const int rl = j + 8 * hl;
    const float inv = 1.0f / lrow[wave][rl];
    const size_t rowg = (size_t)(b * Ldim + q0 + rl) * Ddim + h * DHdim;
    o16[rowg + r]      = (_Float16)(o0[j] * inv);
    o16[rowg + 16 + r] = (_Float16)(o1[j] * inv);
  }
}

// ---------------------------------------------------------------------------
// residual + layernorm: one block per row (D=256 threads)
// ---------------------------------------------------------------------------
__global__ __launch_bounds__(256) void resid_ln(const float* __restrict__ x,
                                                const float* __restrict__ y,
                                                const float* __restrict__ g,
                                                const float* __restrict__ be,
                                                float* __restrict__ out32,
                                                _Float16* __restrict__ out16) {
  __shared__ float buf[256];
  __shared__ float stat;
  const int row = blockIdx.x, t = threadIdx.x;
  const size_t idx = (size_t)row * Ddim + t;
  const float v = x[idx] + y[idx];

  buf[t] = v;
  __syncthreads();
#pragma unroll
  for (int s = 128; s > 0; s >>= 1) {
    if (t < s) buf[t] += buf[t + s];
    __syncthreads();
  }
  if (t == 0) stat = buf[0] * (1.0f / Ddim);
  __syncthreads();
  const float mean = stat;
  const float d = v - mean;
  __syncthreads();
  buf[t] = d * d;
  __syncthreads();
#pragma unroll
  for (int s = 128; s > 0; s >>= 1) {
    if (t < s) buf[t] += buf[t + s];
    __syncthreads();
  }
  if (t == 0) stat = buf[0] * (1.0f / Ddim);
  __syncthreads();
  const float var = stat;

  const float r = d * rsqrtf(var + 1e-5f) * g[t] + be[t];
  out32[idx] = r;
  if (out16) out16[idx] = (_Float16)r;
}

// ---------------------------------------------------------------------------
// launcher
// ---------------------------------------------------------------------------
extern "C" void kernel_launch(void* const* d_in, const int* in_sizes, int n_in,
                              void* d_out, int out_size, void* d_ws, size_t ws_size,
                              hipStream_t stream) {
  const float* x  = (const float*)d_in[0];
  const float* Wq = (const float*)d_in[1];  const float* bq = (const float*)d_in[2];
  const float* Wk = (const float*)d_in[3];  const float* bk = (const float*)d_in[4];
  const float* Wv = (const float*)d_in[5];  const float* bv = (const float*)d_in[6];
  const float* Wo = (const float*)d_in[7];  const float* bo = (const float*)d_in[8];
  const float* W1 = (const float*)d_in[9];  const float* b1 = (const float*)d_in[10];
  const float* W2 = (const float*)d_in[11]; const float* b2 = (const float*)d_in[12];
  const float* g1 = (const float*)d_in[13]; const float* be1 = (const float*)d_in[14];
  const float* g2 = (const float*)d_in[15]; const float* be2 = (const float*)d_in[16];

  char* ws = (char*)d_ws;
  size_t off = 0;
  auto take = [&](size_t bytes) -> void* {
    void* p = ws + off;
    off += (bytes + 255) & ~(size_t)255;
    return p;
  };

  _Float16* x16  = (_Float16*)take((size_t)BLdim * Ddim * 2);
  _Float16* wqt  = (_Float16*)take((size_t)Ddim * Ddim * 2);
  _Float16* wkt  = (_Float16*)take((size_t)Ddim * Ddim * 2);
  _Float16* wvt  = (_Float16*)take((size_t)Ddim * Ddim * 2);
  _Float16* wot  = (_Float16*)take((size_t)Ddim * Ddim * 2);
  _Float16* w1t  = (_Float16*)take((size_t)Ddim * FFdim * 2);
  _Float16* w2t  = (_Float16*)take((size_t)FFdim * Ddim * 2);
  _Float16* qh   = (_Float16*)take((size_t)BLdim * Ddim * 2);
  _Float16* kh   = (_Float16*)take((size_t)BLdim * Ddim * 2);
  _Float16* vt   = (_Float16*)take((size_t)BLdim * Ddim * 2);
  _Float16* o16  = (_Float16*)take((size_t)BLdim * Ddim * 2);
  float*    att32 = (float*)take((size_t)BLdim * Ddim * 4);   // reused as ff2
  float*    h32  = (float*)take((size_t)BLdim * Ddim * 4);
  _Float16* h16  = (_Float16*)take((size_t)BLdim * Ddim * 2);
  _Float16* ff1  = (_Float16*)take((size_t)BLdim * FFdim * 2);
  float*    ff2  = att32;  // alias: att32 dead after LN1

  // 1) f16 conversions
  {
    int n = BLdim * Ddim;
    cvt_f32_to_f16<<<(n + 255) / 256, 256, 0, stream>>>(x, x16, n);
  }
  transpose_cvt<<<(Ddim * Ddim + 255) / 256, 256, 0, stream>>>(Wq, wqt, Ddim, Ddim);
  transpose_cvt<<<(Ddim * Ddim + 255) / 256, 256, 0, stream>>>(Wk, wkt, Ddim, Ddim);
  transpose_cvt<<<(Ddim * Ddim + 255) / 256, 256, 0, stream>>>(Wv, wvt, Ddim, Ddim);
  transpose_cvt<<<(Ddim * Ddim + 255) / 256, 256, 0, stream>>>(Wo, wot, Ddim, Ddim);
  transpose_cvt<<<(Ddim * FFdim + 255) / 256, 256, 0, stream>>>(W1, w1t, Ddim, FFdim);
  transpose_cvt<<<(FFdim * Ddim + 255) / 256, 256, 0, stream>>>(W2, w2t, FFdim, Ddim);

  const float qscale = 0.17677669529663687f;  // 1/sqrt(DH), folded into Q

  // 2) QKV projections -> head layouts
  gemm_wmma<<<dim3(BLdim / GEMM_BM, Ddim / GEMM_BN), 256, 0, stream>>>(
      x16, wqt, bq, qh, BLdim, Ddim, Ddim, MODE_F16_HEADS, qscale);
  gemm_wmma<<<dim3(BLdim / GEMM_BM, Ddim / GEMM_BN), 256, 0, stream>>>(
      x16, wkt, bk, kh, BLdim, Ddim, Ddim, MODE_F16_HEADS, 1.0f);
  gemm_wmma<<<dim3(BLdim / GEMM_BM, Ddim / GEMM_BN), 256, 0, stream>>>(
      x16, wvt, bv, vt, BLdim, Ddim, Ddim, MODE_F16_HEADST, 1.0f);

  // 3) attention
  flash_attn<<<Bdim * Hdim * (Ldim / 64), 128, 0, stream>>>(qh, kh, vt, o16);

  // 4) output projection + residual + LN1
  gemm_wmma<<<dim3(BLdim / GEMM_BM, Ddim / GEMM_BN), 256, 0, stream>>>(
      o16, wot, bo, att32, BLdim, Ddim, Ddim, MODE_F32_PLAIN, 1.0f);
  resid_ln<<<BLdim, 256, 0, stream>>>(x, att32, g1, be1, h32, h16);

  // 5) FFN
  gemm_wmma<<<dim3(BLdim / GEMM_BM, FFdim / GEMM_BN), 256, 0, stream>>>(
      h16, w1t, b1, ff1, BLdim, FFdim, Ddim, MODE_F16_RELU, 1.0f);
  gemm_wmma<<<dim3(BLdim / GEMM_BM, Ddim / GEMM_BN), 256, 0, stream>>>(
      ff1, w2t, b2, ff2, BLdim, Ddim, FFdim, MODE_F32_PLAIN, 1.0f);

  // 6) residual + LN2 -> output
  resid_ln<<<BLdim, 256, 0, stream>>>(h32, ff2, g2, be2, (float*)d_out, nullptr);
}
